// VAE_RNN_4964982194678
// MI455X (gfx1250) — matrix operand order repaired and checked by reading, not compiled, not run on hardware
//
#include <hip/hip_runtime.h>
#include <hip/hip_bf16.h>
#include <math.h>

typedef __attribute__((ext_vector_type(16))) _Float16 v16h;
typedef __attribute__((ext_vector_type(8)))  _Float16 v8h;
typedef __attribute__((ext_vector_type(4)))  _Float16 v4h;
typedef __attribute__((ext_vector_type(2)))  __fp16   v2fp;   // cvt_pkrtz return type
typedef __attribute__((ext_vector_type(8)))  float    v8f;

#define B_    4096
#define T_    200
#define D_    128
#define L_    64
#define H_    128
#define CIN_  256

#define BM    32        // batch rows per block (2 waves x 16 rows)
#define LDH   264       // padded row stride (halves) of the h=[y|s|x] buffer
#define LDA   136       // padded row stride (halves) of the activation buffers

// half-unit offsets inside dynamic LDS
#define OFF_W1U  0              // 256x128 fragment-major
#define OFF_W1R  32768
#define OFF_W1N  65536
#define OFF_W2U  98304          // 128x64
#define OFF_W2R  106496         // 128x64
#define OFF_W2N  114688         // 128x128
#define OFF_H    131072         // BM*LDH
#define OFF_ACT0 139520         // BM*LDA
#define OFF_ACT1 143872         // BM*LDA
#define HALF_END 148224
#define SMEM_BYTES (HALF_END*2 + 640*4)   // 299008 B < 320 KB WGP LDS

// fast elementwise (TRANS ops, no IEEE division sequences)
__device__ __forceinline__ float fast_sigmoid(float x) {
  return __builtin_amdgcn_rcpf(1.0f + __builtin_amdgcn_exp2f(-1.4426950408889634f * x));
}
__device__ __forceinline__ float fast_tanh(float x) {
  // 1 - 2/(1+exp(2x)); saturates correctly: exp2->inf => 1, exp2->0 => -1
  return 1.0f - 2.0f * __builtin_amdgcn_rcpf(1.0f + __builtin_amdgcn_exp2f(2.8853900817779268f * x));
}

// ---- A fragment: 16x32 f16, ISA layout (lanes 0-15: M=lane, K halves
// [k0..k0+7],[k0+16..k0+23]; lanes 16-31: M=lane-16, K +8) ----
__device__ __forceinline__ v16h load_a_frag(const _Float16* base, int ld, int k0, int lane) {
  const int m    = lane & 15;
  const int koff = (lane < 16) ? 0 : 8;
  const _Float16* p = base + m * ld + k0 + koff;
  v8h lo = *(const v8h*)p;          // ds_load_b128
  v8h hi = *(const v8h*)(p + 16);   // ds_load_b128
  v16h r;
#pragma unroll
  for (int i = 0; i < 8; ++i) { r[i] = lo[i]; r[i + 8] = hi[i]; }
  return r;
}

// ---- B fragment: 32x16 f16, fragment-major: 16 contiguous halves per lane ----
__device__ __forceinline__ v16h load_b_frag(const _Float16* wfrag, int tile, int lane) {
  const _Float16* p = wfrag + tile * 512 + lane * 16;
  v8h lo = *(const v8h*)p;
  v8h hi = *(const v8h*)(p + 8);
  v16h r;
#pragma unroll
  for (int i = 0; i < 8; ++i) { r[i] = lo[i]; r[i + 8] = hi[i]; }
  return r;
}

// ---- single GEMM: [16 x NT*16] = A[16 x KT*32] @ W[KT*32 x NT*16] ----
template<int KT, int NT>
__device__ __forceinline__ void gemm(const _Float16* aBase, int lda,
                                     const _Float16* wFrag, v8f (&acc)[NT], int lane) {
#pragma unroll
  for (int nt = 0; nt < NT; ++nt)
#pragma unroll
    for (int j = 0; j < 8; ++j) acc[nt][j] = 0.0f;
#pragma unroll
  for (int kt = 0; kt < KT; ++kt) {
    v16h a = load_a_frag(aBase, lda, kt * 32, lane);
#pragma unroll
    for (int nt = 0; nt < NT; ++nt) {
      v16h b = load_b_frag(wFrag, nt * KT + kt, lane);
      acc[nt] = __builtin_amdgcn_wmma_f32_16x16x32_f16(
          false, a, false, b, (short)0, acc[nt], false, false);
    }
  }
}

// ---- dual GEMM sharing one A operand (ug & rg first layers):
//      two independent accumulator chains interleaved to hide WMMA hazards ----
template<int KT, int NT>
__device__ __forceinline__ void gemm_dualA(const _Float16* aBase, int lda,
                                           const _Float16* w0, const _Float16* w1,
                                           v8f (&acc0)[NT], v8f (&acc1)[NT], int lane) {
#pragma unroll
  for (int nt = 0; nt < NT; ++nt)
#pragma unroll
    for (int j = 0; j < 8; ++j) { acc0[nt][j] = 0.0f; acc1[nt][j] = 0.0f; }
#pragma unroll
  for (int kt = 0; kt < KT; ++kt) {
    v16h a = load_a_frag(aBase, lda, kt * 32, lane);
#pragma unroll
    for (int nt = 0; nt < NT; ++nt) {
      v16h b0 = load_b_frag(w0, nt * KT + kt, lane);
      v16h b1 = load_b_frag(w1, nt * KT + kt, lane);
      acc0[nt] = __builtin_amdgcn_wmma_f32_16x16x32_f16(
          false, a, false, b0, (short)0, acc0[nt], false, false);
      acc1[nt] = __builtin_amdgcn_wmma_f32_16x16x32_f16(
          false, a, false, b1, (short)0, acc1[nt], false, false);
    }
  }
}

// ---- pair of independent GEMMs (different A and W) interleaved ----
template<int KT, int NT>
__device__ __forceinline__ void gemm_pair(const _Float16* a0, const _Float16* a1, int lda,
                                          const _Float16* w0, const _Float16* w1,
                                          v8f (&acc0)[NT], v8f (&acc1)[NT], int lane) {
#pragma unroll
  for (int nt = 0; nt < NT; ++nt)
#pragma unroll
    for (int j = 0; j < 8; ++j) { acc0[nt][j] = 0.0f; acc1[nt][j] = 0.0f; }
#pragma unroll
  for (int kt = 0; kt < KT; ++kt) {
    v16h fa0 = load_a_frag(a0, lda, kt * 32, lane);
    v16h fa1 = load_a_frag(a1, lda, kt * 32, lane);
#pragma unroll
    for (int nt = 0; nt < NT; ++nt) {
      v16h b0 = load_b_frag(w0, nt * KT + kt, lane);
      v16h b1 = load_b_frag(w1, nt * KT + kt, lane);
      acc0[nt] = __builtin_amdgcn_wmma_f32_16x16x32_f16(
          false, fa0, false, b0, (short)0, acc0[nt], false, false);
      acc1[nt] = __builtin_amdgcn_wmma_f32_16x16x32_f16(
          false, fa1, false, b1, (short)0, acc1[nt], false, false);
    }
  }
}

// stage f32 row-major [K][N] weight -> f16 fragment-major LDS
__device__ __forceinline__ void stage_w(const float* __restrict__ g, _Float16* dst,
                                        int K, int N, int tid, int nthreads) {
  const int numKt = K >> 5;
  for (int idx = tid; idx < K * N; idx += nthreads) {
    int k = idx / N, n = idx - k * N;
    int kt = k >> 5, kk = k & 31, nt = n >> 4, nn = n & 15;
    int lane = (((kk >> 3) & 1) << 4) | nn;
    int i = (kk & 7) | ((kk >> 4) << 3);
    dst[(nt * numKt + kt) * 512 + lane * 16 + i] = (_Float16)g[idx];
  }
}

__global__ void __launch_bounds__(64) vae_rnn_kernel(
    const float* __restrict__ data,
    const float* __restrict__ ug_w1, const float* __restrict__ ug_b1,
    const float* __restrict__ ug_w2, const float* __restrict__ ug_b2,
    const float* __restrict__ rg_w1, const float* __restrict__ rg_b1,
    const float* __restrict__ rg_w2, const float* __restrict__ rg_b2,
    const float* __restrict__ ns_w1, const float* __restrict__ ns_b1,
    const float* __restrict__ ns_w2, const float* __restrict__ ns_b2,
    float* __restrict__ out)
{
  extern __shared__ __align__(16) char smem_raw[];
  _Float16* sm = (_Float16*)smem_raw;
  float*    sf = (float*)(smem_raw + HALF_END * 2);

  const int tid  = threadIdx.x;
  const int lane = tid & 31;
  const int wave = tid >> 5;
  const int nthr = blockDim.x;   // 64

  // ---- one-time weight staging (f32 global -> f16 fragment-major LDS) ----
  stage_w(ug_w1, sm + OFF_W1U, CIN_, H_, tid, nthr);
  stage_w(rg_w1, sm + OFF_W1R, CIN_, H_, tid, nthr);
  stage_w(ns_w1, sm + OFF_W1N, CIN_, H_, tid, nthr);
  stage_w(ug_w2, sm + OFF_W2U, H_, L_,     tid, nthr);
  stage_w(rg_w2, sm + OFF_W2R, H_, L_,     tid, nthr);
  stage_w(ns_w2, sm + OFF_W2N, H_, 2 * L_, tid, nthr);
  for (int i = tid; i < 128; i += nthr) {
    sf[i] = ug_b1[i]; sf[128 + i] = rg_b1[i]; sf[256 + i] = ns_b1[i]; sf[512 + i] = ns_b2[i];
  }
  for (int i = tid; i < 64; i += nthr) { sf[384 + i] = ug_b2[i]; sf[448 + i] = rg_b2[i]; }
  // zero initial state y0,s0 (h cols 0..127)
  for (int e = tid; e < BM * 128; e += nthr)
    sm[OFF_H + (e >> 7) * LDH + (e & 127)] = (_Float16)0.0f;
  __syncthreads();

  const int wr     = wave * 16;            // wave's first row in the block tile
  const int grBase = blockIdx.x * BM + wr; // first global batch row for this wave
  _Float16* hBase  = sm + OFF_H    + wr * LDH;
  _Float16* a0Base = sm + OFF_ACT0 + wr * LDA;
  _Float16* a1Base = sm + OFF_ACT1 + wr * LDA;

  const int ncol = lane & 15;              // C/D layout: N = nt*16 + ncol
  const int mb   = (lane < 16) ? 0 : 8;    //            M = mb + j

  // ---- biases preloaded into persistent registers (per-lane column) ----
  float b1uR[8], b1rR[8], b1nR[8], b2nR[8], b2uR[4], b2rR[4];
#pragma unroll
  for (int nt = 0; nt < 8; ++nt) {
    int n = nt * 16 + ncol;
    b1uR[nt] = sf[n]; b1rR[nt] = sf[128 + n]; b1nR[nt] = sf[256 + n]; b2nR[nt] = sf[512 + n];
  }
#pragma unroll
  for (int nt = 0; nt < 4; ++nt) {
    int n = nt * 16 + ncol;
    b2uR[nt] = sf[384 + n]; b2rR[nt] = sf[448 + n];
  }

  // ---- y,s state held in registers (C/D layout) for the whole recurrence ----
  float yreg[4][8], sreg[4][8];
#pragma unroll
  for (int nt = 0; nt < 4; ++nt)
#pragma unroll
    for (int j = 0; j < 8; ++j) { yreg[nt][j] = 0.0f; sreg[nt][j] = 0.0f; }

  for (int t = 0; t < T_; ++t) {
    // ---- x tile: 16 rows x 128 f32 -> f16 into h cols 128..255 ----
#pragma unroll 4
    for (int e = lane; e < 16 * 32; e += 32) {
      int r = e >> 5, c4 = e & 31;
      const float4 v = *(const float4*)(data + (size_t)(grBase + r) * (T_ * D_)
                                             + (size_t)t * D_ + c4 * 4);
      v2fp lo = __builtin_amdgcn_cvt_pkrtz(v.x, v.y);   // v_cvt_pkrtz_f16_f32
      v2fp hi = __builtin_amdgcn_cvt_pkrtz(v.z, v.w);
      v4h h4;
      h4[0] = (_Float16)lo[0]; h4[1] = (_Float16)lo[1];
      h4[2] = (_Float16)hi[0]; h4[3] = (_Float16)hi[1];
      *(v4h*)(hBase + r * LDH + 128 + c4 * 4) = h4;
    }
    // prefetch next step's rows into GL2
    if (t + 1 < T_) {
      int pr = lane & 15;
      __builtin_prefetch(data + (size_t)(grBase + pr) * (T_ * D_)
                              + (size_t)(t + 1) * D_ + (lane >> 4) * 64, 0, 1);
    }

    v8f g1a[8], g1b[8], g2a[4], g2b[4], g3[8];
    float u[4][8], rc[4][8];

    // ---- layer-1 for u and r gates: shared A, dual accumulators ----
    gemm_dualA<8, 8>(hBase, LDH, sm + OFF_W1U, sm + OFF_W1R, g1a, g1b, lane);
#pragma unroll
    for (int nt = 0; nt < 8; ++nt) {
#pragma unroll
      for (int j = 0; j < 8; ++j) {
        int off = (mb + j) * LDA + nt * 16 + ncol;
        a0Base[off] = (_Float16)fast_tanh(g1a[nt][j] + b1uR[nt]);
        a1Base[off] = (_Float16)fast_tanh(g1b[nt][j] + b1rR[nt]);
      }
    }
    // ---- layer-2 for u and r: independent pair, interleaved ----
    gemm_pair<4, 4>(a0Base, a1Base, LDA, sm + OFF_W2U, sm + OFF_W2R, g2a, g2b, lane);
#pragma unroll
    for (int nt = 0; nt < 4; ++nt)
#pragma unroll
      for (int j = 0; j < 8; ++j) {
        u[nt][j]  = fast_sigmoid(g2a[nt][j] + b2uR[nt]);
        rc[nt][j] = fast_sigmoid(g2b[nt][j] + b2rR[nt]);
      }

    // ---- hc = [y*r | s*r | x] (overwrite h cols 0..127) ----
#pragma unroll
    for (int nt = 0; nt < 4; ++nt) { int n = nt * 16 + ncol;
#pragma unroll
      for (int j = 0; j < 8; ++j) {
        hBase[(mb + j) * LDH + n]      = (_Float16)(yreg[nt][j] * rc[nt][j]);
        hBase[(mb + j) * LDH + 64 + n] = (_Float16)(sreg[nt][j] * rc[nt][j]);
      }
    }

    // ---- ns = tanh(hc@W1n+b1n)@W2n+b2n ; state update ----
    gemm<8, 8>(hBase, LDH, sm + OFF_W1N, g1a, lane);
#pragma unroll
    for (int nt = 0; nt < 8; ++nt)
#pragma unroll
      for (int j = 0; j < 8; ++j)
        a0Base[(mb + j) * LDA + nt * 16 + ncol] = (_Float16)fast_tanh(g1a[nt][j] + b1nR[nt]);
    gemm<4, 8>(a0Base, LDA, sm + OFF_W2N, g3, lane);

#pragma unroll
    for (int nt = 0; nt < 4; ++nt) { int n = nt * 16 + ncol;
#pragma unroll
      for (int j = 0; j < 8; ++j) {
        float uu   = u[nt][j];
        float mean = g3[nt][j]     + b2nR[nt];
        float sd   = fabsf(g3[nt + 4][j] + b2nR[nt + 4]);
        float y2   = (1.0f - uu) * mean + uu * yreg[nt][j];
        float s2   = (1.0f - uu) * sd   + uu * sreg[nt][j];
        yreg[nt][j] = y2; sreg[nt][j] = s2;
        hBase[(mb + j) * LDH + n]      = (_Float16)y2;   // f16 copy for next step's GEMM
        hBase[(mb + j) * LDH + 64 + n] = (_Float16)s2;
      }
    }
  }

  // ---- write yT, sT (f32) straight from register state ----
#pragma unroll
  for (int nt = 0; nt < 4; ++nt) { int n = nt * 16 + ncol;
#pragma unroll
    for (int j = 0; j < 8; ++j) {
      size_t row = (size_t)(grBase + mb + j);
      out[row * L_ + n]                   = yreg[nt][j];
      out[(size_t)B_ * L_ + row * L_ + n] = sreg[nt][j];
    }
  }
}

extern "C" void kernel_launch(void* const* d_in, const int* in_sizes, int n_in,
                              void* d_out, int out_size, void* d_ws, size_t ws_size,
                              hipStream_t stream) {
  const float* data  = (const float*)d_in[0];
  // d_in[1] = time_steps (unused by the reference computation)
  const float* ug_w1 = (const float*)d_in[2];  const float* ug_b1 = (const float*)d_in[3];
  const float* ug_w2 = (const float*)d_in[4];  const float* ug_b2 = (const float*)d_in[5];
  const float* rg_w1 = (const float*)d_in[6];  const float* rg_b1 = (const float*)d_in[7];
  const float* rg_w2 = (const float*)d_in[8];  const float* rg_b2 = (const float*)d_in[9];
  const float* ns_w1 = (const float*)d_in[10]; const float* ns_b1 = (const float*)d_in[11];
  const float* ns_w2 = (const float*)d_in[12]; const float* ns_b2 = (const float*)d_in[13];
  float* out = (float*)d_out;

  dim3 grid(B_ / BM), block(64);
  hipLaunchKernelGGL(vae_rnn_kernel, grid, block, SMEM_BYTES, stream,
                     data,
                     ug_w1, ug_b1, ug_w2, ug_b2,
                     rg_w1, rg_b1, rg_w2, rg_b2,
                     ns_w1, ns_b1, ns_w2, ns_b2,
                     out);
}
